// SentenceEnforcedSwitchMoeBlock_82197084111405
// MI455X (gfx1250) — matrix operand order repaired and checked by reading, not compiled, not run on hardware
//
#include <hip/hip_runtime.h>
#include <hip/hip_bf16.h>
#include <stdint.h>

typedef __attribute__((ext_vector_type(16))) __bf16 v16bf;
typedef __attribute__((ext_vector_type(8)))  float  v8f;

#define B_  64
#define S_  2048
#define D_  512
#define F_  2048
#define E_  8

#define MT  16            // S-rows per workgroup
#define KT1 (D_/32)       // 16 k-tiles, GEMM1
#define NT1 (F_/16)       // 128 n-tiles, GEMM1
#define KT2 (F_/32)       // 64 k-tiles, GEMM2
#define NT2 (D_/16)       // 32 n-tiles, GEMM2

#define XST (D_ + 8)      // LDS stride (bf16 elems) for X tile -> conflict-free b128
#define HST (F_ + 8)      // LDS stride (bf16 elems) for H tile

__device__ __forceinline__ unsigned f2bf_pack(float a, float b) {
  union { float f; unsigned u; } ca, cb;
  ca.f = a; cb.f = b;
  unsigned ra = ca.u + 0x7FFFu + ((ca.u >> 16) & 1u);   // RNE
  unsigned rb = cb.u + 0x7FFFu + ((cb.u >> 16) & 1u);
  return (ra >> 16) | (rb & 0xFFFF0000u);
}

__device__ __forceinline__ unsigned short f2bf(float a) {
  union { float f; unsigned u; } c; c.f = a;
  unsigned r = c.u + 0x7FFFu + ((c.u >> 16) & 1u);
  return (unsigned short)(r >> 16);
}

// Branchless tanh-approx GELU (jax.nn.gelu approximate=True).
// tanh via hardware exp2 (v_exp_f32) + rcp (v_rcp_f32); sign via v_bfi_b32.
__device__ __forceinline__ float fast_gelu(float x) {
  float x3 = x * x * x;
  float y  = 0.7978845608028654f * (x + 0.044715f * x3);
  float ay = __builtin_fabsf(y);
  float u  = __builtin_amdgcn_exp2f(-2.885390081777927f * ay); // 2^(-2*log2e*|y|)
  float th = (1.0f - u) * __builtin_amdgcn_rcpf(1.0f + u);     // tanh(|y|)
  th = __builtin_copysignf(th, y);
  return 0.5f * x * (1.0f + th);
}

// Pre-swizzle fp32 weights (E, Dk, Fn) into WMMA bf16 B-fragment layout:
// dst[((e*Nt + nt)*Kt + kt)*32*8 + lane*8 + v] = pack(bf16 W[k][n], bf16 W[k+1][n])
// with k = kt*32 + 16*(lane>>4) + 2*v, n = nt*16 + (lane&15)
__global__ __launch_bounds__(256)
void swizzle_w_kernel(const float* __restrict__ W, unsigned* __restrict__ dst,
                      int Dk, int Fn) {
  int Kt = Dk / 32, Nt = Fn / 16;
  long long total = (long long)E_ * Nt * Kt * 256;
  long long i = (long long)blockIdx.x * blockDim.x + threadIdx.x;
  if (i >= total) return;
  int v    = (int)(i & 7);
  int lane = (int)((i >> 3) & 31);
  long long t = i >> 8;
  int kt = (int)(t % Kt);  t /= Kt;
  int nt = (int)(t % Nt);  t /= Nt;
  int e  = (int)t;
  int k = kt * 32 + ((lane >> 4) << 4) + 2 * v;
  int n = nt * 16 + (lane & 15);
  const float* w = W + ((long long)e * Dk + k) * Fn + n;
  dst[i] = f2bf_pack(w[0], w[Fn]);
}

// Fused per-(s-tile, b) MoE FFN: X@W1[e] + b1 -> GELU -> @W2[e] + b2
__global__ __launch_bounds__(256)
void moe_ffn_kernel(const float* __restrict__ X, const int* __restrict__ assign,
                    const unsigned* __restrict__ W1s, const float* __restrict__ b1,
                    const unsigned* __restrict__ W2s, const float* __restrict__ b2,
                    float* __restrict__ Out) {
  extern __shared__ char smem[];
  unsigned short* Xs = (unsigned short*)smem;                          // MT x XST bf16
  unsigned short* Hs = (unsigned short*)(smem + (size_t)MT * XST * 2); // MT x HST bf16

  const int sTile = blockIdx.x;          // 0..S/MT-1
  const int b     = blockIdx.y;          // 0..B-1
  const int e     = assign[b];
  const int s0    = sTile * MT;

  const int tid  = threadIdx.x;
  const int lane = tid & 31;
  const int wave = tid >> 5;
  const int m    = lane & 15;            // A row / C-D column index
  const int kh   = lane >> 4;            // K/M half selector

  // ---- stage X tile (16 x 512 fp32) into LDS as bf16 ----
  const float* xg = X + ((long long)b * S_ + s0) * D_;
  for (int i = tid; i < MT * (D_ / 2); i += 256) {
    int r = i >> 8;                      // D_/2 == 256
    int c = (i & 255) * 2;
    float2 xv = *(const float2*)(xg + (long long)r * D_ + c);
    *(unsigned*)(Xs + r * XST + c) = f2bf_pack(xv.x, xv.y);
  }
  __syncthreads();

  // ---- GEMM1: (16x512) x (512x2048) + bias -> GELU -> Hs ----
  const unsigned* w1e = W1s + (long long)e * NT1 * KT1 * 256;
  for (int t = wave; t < NT1; t += 8) {
    const int n0 = t * 16;
    const float bias = b1[e * F_ + n0 + m];
    v8f acc = {};
    const unsigned* wb = w1e + (long long)t * KT1 * 256 + lane * 8;
    #pragma unroll 4
    for (int kt = 0; kt < KT1; ++kt) {
      union { uint4 u[2]; v16bf v; } a, bm;
      const unsigned short* ap = Xs + m * XST + kt * 32 + kh * 16;
      a.u[0]  = *(const uint4*)(ap);        // VGPR0-3: K = 16*kh + 0..7
      a.u[1]  = *(const uint4*)(ap + 8);    // VGPR4-7: K = 16*kh + 8..15
      bm.u[0] = *(const uint4*)(wb);
      bm.u[1] = *(const uint4*)(wb + 4);
      wb += 256;
      __builtin_prefetch(wb, 0, 1);         // global_prefetch_b8 next k-tile
      acc = __builtin_amdgcn_wmma_f32_16x16x32_bf16(
          false, a.v, false, bm.v, (short)0, acc, false, false);
    }
    #pragma unroll
    for (int r = 0; r < 8; ++r) {
      Hs[(r + 8 * kh) * HST + n0 + m] = f2bf(fast_gelu(acc[r] + bias));
    }
  }
  __syncthreads();

  // ---- GEMM2: (16x2048) x (2048x512) + bias -> Out ----
  const unsigned* w2e = W2s + (long long)e * NT2 * KT2 * 256;
  for (int t = wave; t < NT2; t += 8) {
    const int n0 = t * 16;
    const float bias = b2[e * D_ + n0 + m];
    v8f acc = {};
    const unsigned* wb = w2e + (long long)t * KT2 * 256 + lane * 8;
    #pragma unroll 4
    for (int kt = 0; kt < KT2; ++kt) {
      union { uint4 u[2]; v16bf v; } a, bm;
      const unsigned short* ap = Hs + m * HST + kt * 32 + kh * 16;
      a.u[0]  = *(const uint4*)(ap);
      a.u[1]  = *(const uint4*)(ap + 8);
      bm.u[0] = *(const uint4*)(wb);
      bm.u[1] = *(const uint4*)(wb + 4);
      wb += 256;
      __builtin_prefetch(wb, 0, 1);
      acc = __builtin_amdgcn_wmma_f32_16x16x32_bf16(
          false, a.v, false, bm.v, (short)0, acc, false, false);
    }
    float* og = Out + ((long long)b * S_ + s0) * D_ + n0 + m;
    #pragma unroll
    for (int r = 0; r < 8; ++r) {
      og[(long long)(r + 8 * kh) * D_] = acc[r] + bias;
    }
  }
}

extern "C" void kernel_launch(void* const* d_in, const int* in_sizes, int n_in,
                              void* d_out, int out_size, void* d_ws, size_t ws_size,
                              hipStream_t stream) {
  const float* X      = (const float*)d_in[0];
  const int*   assign = (const int*)  d_in[1];
  const float* W1     = (const float*)d_in[2];
  const float* b1     = (const float*)d_in[3];
  const float* W2     = (const float*)d_in[4];
  const float* b2     = (const float*)d_in[5];
  float* Out = (float*)d_out;

  unsigned* W1s = (unsigned*)d_ws;                      // E*NT1*KT1*256 u32 = 16.8 MB
  size_t w1n = (size_t)E_ * NT1 * KT1 * 256;
  unsigned* W2s = W1s + w1n;                            // + 16.8 MB
  size_t w2n = (size_t)E_ * NT2 * KT2 * 256;

  const int thr = 256;
  swizzle_w_kernel<<<(unsigned)((w1n + thr - 1) / thr), thr, 0, stream>>>(W1, W1s, D_, F_);
  swizzle_w_kernel<<<(unsigned)((w2n + thr - 1) / thr), thr, 0, stream>>>(W2, W2s, F_, D_);

  dim3 grid(S_ / MT, B_);
  size_t shmem = (size_t)MT * (XST + HST) * 2;          // 82,432 B < 320 KB WGP LDS
  moe_ffn_kernel<<<grid, 256, shmem, stream>>>(X, assign, W1s, b1, W2s, b2, Out);
}
